// BiBayesianConv_70489003262724
// MI455X (gfx1250) — matrix operand order repaired and checked by reading, not compiled, not run on hardware
//
#include <hip/hip_runtime.h>

// ---------------------------------------------------------------------------
// BiBayesianConv on MI455X (gfx1250).
// Reduces to: Weff[f,c] = sum_kl tanh(w+0.5*log(eps/(1-eps))), then
// out[b,f,p] = sum_c Weff[f,c] * x[b,c,p]  (GEMM, M=512,K=256,N=64*4096).
// Memory-bound (~0.81GB @ 23.3TB/s => ~35us). bf16 WMMA, f32 accumulate,
// split precision: xh*wh + xl*wh + xh*wl  (~2^-16 rel err vs fp32).
// This round: v_perm_b32-based bf16 pair packing (cuts VALU ~25%) and
// global_prefetch_b8 of the next K-slab of x.
// ---------------------------------------------------------------------------

typedef __attribute__((ext_vector_type(16))) __bf16 v16bf;
typedef __attribute__((ext_vector_type(8)))  float  v8f;

#define BB 64
#define CC 256
#define FF 512
#define WW 64
#define HH 64
#define PP (WW * HH)   /* 4096 pixels, contiguous per (b,c) */
#define KK 9           /* 3x3 kernel positions */

// pack {trunc_bf16(f1), trunc_bf16(f0)} into one dword with one v_perm_b32
__device__ __forceinline__ unsigned pack_bf16_trunc(float f0, float f1) {
    return __builtin_amdgcn_perm(__float_as_uint(f1), __float_as_uint(f0),
                                 0x07060302u);
}

// ---------------------------------------------------------------------------
// Phase 1: Weff[f,c] = sum_{i<9} tanh(w[i] + 0.5*log(e/(1-e)))  (TAU = 1)
// Store split bf16: whi = trunc_bf16(Weff), wlo = trunc_bf16(Weff - whi),
// row-major [F, C] so WMMA A-fragments are 16B-contiguous loads.
// ---------------------------------------------------------------------------
__global__ __launch_bounds__(256) void weff_kernel(
    const float* __restrict__ weight, const float* __restrict__ eps,
    unsigned short* __restrict__ whi, unsigned short* __restrict__ wlo)
{
    int idx = blockIdx.x * blockDim.x + threadIdx.x;   // idx = f*C + c
    if (idx >= FF * CC) return;
    const float* wp = weight + (size_t)idx * KK;
    const float* ep = eps    + (size_t)idx * KK;       // S==1: same flat layout
    float s = 0.0f;
#pragma unroll
    for (int i = 0; i < KK; ++i) {
        float e     = ep[i];
        float delta = 0.5f * logf(e / (1.0f - e));
        s += tanhf(wp[i] + delta);
    }
    unsigned uh = __float_as_uint(s) & 0xFFFF0000u;
    float    r  = s - __uint_as_float(uh);
    whi[idx] = (unsigned short)(uh >> 16);
    wlo[idx] = (unsigned short)(__float_as_uint(r) >> 16);
}

// ---------------------------------------------------------------------------
// Phase 2: out[b,f,p] = sum_c Weff[f,c]*x[b,c,p] via v_wmma_f32_16x16x32_bf16.
// Workgroup = 8 waves, owns (b, 32-pixel strip); 4 f-chunks x 8 waves x 16
// rows = all 512 f, so x is read from HBM exactly once.
// ---------------------------------------------------------------------------
__global__ __launch_bounds__(256) void gemm_kernel(
    const float* __restrict__ x,
    const unsigned short* __restrict__ whi,
    const unsigned short* __restrict__ wlo,
    float* __restrict__ out)
{
    const int b     = blockIdx.y;
    const int p0    = blockIdx.x * 32;       // 2 tiles of 16 pixels (128B aligned)
    const int wave  = threadIdx.x >> 5;      // 0..7 (wave32)
    const int lane  = threadIdx.x & 31;
    const int m     = lane & 15;
    const int khalf = lane >> 4;             // 0: low K half-lanes, 1: high

    v8f acc[4][2];
#pragma unroll
    for (int ch = 0; ch < 4; ++ch)
#pragma unroll
        for (int t = 0; t < 2; ++t)
            acc[ch][t] = (v8f){0.f,0.f,0.f,0.f,0.f,0.f,0.f,0.f};

    const float* xb = x + (size_t)b * CC * PP;

#pragma unroll 1
    for (int c0 = 0; c0 < CC; c0 += 32) {
        // prefetch next K-slab: lane L covers row c0+32+L at the strip start
        // (each strip row is one 128B line) -> global_prefetch_b8
        if (c0 + 32 < CC) {
            __builtin_prefetch(xb + (size_t)(c0 + 32 + lane) * PP + p0, 0, 2);
        }

        // --- B fragments: 32(K=c) x 16(N=p) bf16, hi/lo split --------------
        // lane n<16: col p0+t*16+n, K=0..15 ; lane n+16: same col, K=16..31
        union { unsigned u[8]; v16bf v; } bhi[2], blo[2];
        const int koff = khalf * 16;
#pragma unroll
        for (int t = 0; t < 2; ++t) {
            const float* xp = xb + (size_t)(c0 + koff) * PP + (p0 + t * 16 + m);
#pragma unroll
            for (int kk = 0; kk < 8; ++kk) {
                float f0 = xp[(size_t)(2 * kk    ) * PP];
                float f1 = xp[(size_t)(2 * kk + 1) * PP];
                bhi[t].u[kk] = pack_bf16_trunc(f0, f1);
                float h0 = __uint_as_float(__float_as_uint(f0) & 0xFFFF0000u);
                float h1 = __uint_as_float(__float_as_uint(f1) & 0xFFFF0000u);
                blo[t].u[kk] = pack_bf16_trunc(f0 - h0, f1 - h1);
            }
        }

        // --- A fragments + WMMA for 4 f-chunks -----------------------------
        // A 16x32 in this code's contiguous k order (same as B): lane row m, K = 16*khalf + 0..15
#pragma unroll
        for (int ch = 0; ch < 4; ++ch) {
            const int fr = ch * 128 + wave * 16 + m;
            union { uint4 q[2]; v16bf v; } ahi, alo;
            const unsigned short* ph = whi + (size_t)fr * CC + c0 + khalf * 16;
            const unsigned short* pl = wlo + (size_t)fr * CC + c0 + khalf * 16;
            ahi.q[0] = *reinterpret_cast<const uint4*>(ph);
            ahi.q[1] = *reinterpret_cast<const uint4*>(ph + 8);
            alo.q[0] = *reinterpret_cast<const uint4*>(pl);
            alo.q[1] = *reinterpret_cast<const uint4*>(pl + 8);
#pragma unroll
            for (int t = 0; t < 2; ++t) {
                acc[ch][t] = __builtin_amdgcn_wmma_f32_16x16x32_bf16(
                    false, ahi.v, false, bhi[t].v, (short)0, acc[ch][t], false, false);
                acc[ch][t] = __builtin_amdgcn_wmma_f32_16x16x32_bf16(
                    false, alo.v, false, bhi[t].v, (short)0, acc[ch][t], false, false);
                acc[ch][t] = __builtin_amdgcn_wmma_f32_16x16x32_bf16(
                    false, ahi.v, false, blo[t].v, (short)0, acc[ch][t], false, false);
            }
        }
    }

    // --- store: D VGPR r -> (row f+r, cols p0..p0+15) / (row f+8+r) --------
    // 64B contiguous per half-wave; tile t=0/1 halves combine to full lines.
#pragma unroll
    for (int ch = 0; ch < 4; ++ch) {
        const int fbase = ch * 128 + wave * 16;
#pragma unroll
        for (int t = 0; t < 2; ++t) {
#pragma unroll
            for (int r = 0; r < 8; ++r) {
                int frow = fbase + khalf * 8 + r;
                out[((size_t)b * FF + frow) * PP + (p0 + t * 16 + m)] = acc[ch][t][r];
            }
        }
    }
}

// ---------------------------------------------------------------------------
extern "C" void kernel_launch(void* const* d_in, const int* in_sizes, int n_in,
                              void* d_out, int out_size, void* d_ws, size_t ws_size,
                              hipStream_t stream) {
    const float* x      = (const float*)d_in[0];
    const float* weight = (const float*)d_in[1];
    const float* eps    = (const float*)d_in[2];
    float* out = (float*)d_out;

    unsigned short* whi = (unsigned short*)d_ws;         // 512*256 bf16
    unsigned short* wlo = whi + (size_t)FF * CC;         // 512*256 bf16

    weff_kernel<<<(FF * CC) / 256, 256, 0, stream>>>(weight, eps, whi, wlo);
    gemm_kernel<<<dim3(PP / 32, BB), 256, 0, stream>>>(x, whi, wlo, out);
}
